// FraudRGCN_13108240187667
// MI455X (gfx1250) — matrix operand order repaired and checked by reading, not compiled, not run on hardware
//
#include <hip/hip_runtime.h>
#include <stdint.h>

// Problem constants (from reference)
#define NN 100000
#define NE 1600000
#define NR 4
// A matrix layout: per node row of 640 bf16: [agg_r0(128) agg_r1 agg_r2 agg_r3 | x(128)]
#define AST 640

typedef __attribute__((ext_vector_type(16))) __bf16 v16bf;
typedef __attribute__((ext_vector_type(8)))  float  v8f;
union FragBF { uint4 q[2]; v16bf v; };

__device__ __forceinline__ unsigned short f32_bf16(float f) {
  unsigned u = __float_as_uint(f);
  u += 0x7FFFu + ((u >> 16) & 1u);           // round-to-nearest-even
  return (unsigned short)(u >> 16);
}
__device__ __forceinline__ float bf16_f32(unsigned short h) {
  return __uint_as_float(((unsigned)h) << 16);
}

// ---------------- zero ----------------
__global__ void k_zero_u4(uint4* p, long n) {
  long stride = (long)gridDim.x * blockDim.x;
  uint4 z = {0u, 0u, 0u, 0u};
  for (long i = (long)blockIdx.x * blockDim.x + threadIdx.x; i < n; i += stride) p[i] = z;
}

// ---------------- weight prep: WcatT[o][k] = stacked [W1;root1]^T in bf16 ----------------
__global__ void k_prep_w1(const float* __restrict__ W1, const float* __restrict__ root1,
                          unsigned short* __restrict__ WcatT) {
  int idx = blockIdx.x * blockDim.x + threadIdx.x;   // 640*128 threads
  if (idx >= 640 * 128) return;
  int k = idx >> 7, o = idx & 127;
  float v = (k < 512) ? W1[idx]                       // W1 flat [(r*128+d)*128+o] == [k*128+o]
                      : root1[((k - 512) << 7) + o];
  WcatT[o * AST + k] = f32_bf16(v);
}

// Wc2T[o][k], o in [0,16): o=2r+c -> W2[r][k][c]; o=8+c -> root2[k][c]; o>=10 -> 0
__global__ void k_prep_w2(const float* __restrict__ W2, const float* __restrict__ root2,
                          unsigned short* __restrict__ Wc2T) {
  int idx = blockIdx.x * blockDim.x + threadIdx.x;   // 16*128 threads
  if (idx >= 16 * 128) return;
  int o = idx >> 7, k = idx & 127;
  float v = 0.0f;
  if (o < 8)       v = W2[(((o >> 1) << 7) + k) * 2 + (o & 1)];
  else if (o < 10) v = root2[k * 2 + (o & 1)];
  Wc2T[o * 128 + k] = f32_bf16(v);
}

// ---------------- x f32 -> bf16 into A's x-region ----------------
__global__ void k_convert_x(const float* __restrict__ x, unsigned short* __restrict__ A) {
  long tid = (long)blockIdx.x * blockDim.x + threadIdx.x;   // NN*64 threads (dword each)
  if (tid >= (long)NN * 64) return;
  long n = tid >> 6; int p = (int)(tid & 63);
  float2 f = *(const float2*)(x + n * 128 + p * 2);
  unsigned lo = f32_bf16(f.x), hi = f32_bf16(f.y);
  ((unsigned*)A)[n * (AST / 2) + 256 + p] = lo | (hi << 16);
}

// ---------------- layer-1 edge scatter: one wave per edge, pk bf16 atomics ----------------
__global__ void k_scatter1(const int* __restrict__ ei, const int* __restrict__ et,
                           unsigned short* __restrict__ A, unsigned* __restrict__ cnt) {
  int lane = threadIdx.x & 31;
  long wid = ((long)blockIdx.x * blockDim.x + threadIdx.x) >> 5;
  long nw  = ((long)gridDim.x * blockDim.x) >> 5;
  unsigned long long abase = (unsigned long long)(uintptr_t)A;
  for (long e = wid; e < NE; e += nw) {
    __builtin_prefetch(ei + e + nw, 0, 1);           // -> global_prefetch_b8
    __builtin_prefetch(ei + NE + e + nw, 0, 1);
    int src = ei[e];
    int dst = ei[NE + e];
    int rel = et[e];
    // gather 8 bytes (4 bf16) of x from A's x-region
    uint2 d = *(const uint2*)(A + (long)src * AST + 512 + lane * 4);
    unsigned off = (unsigned)(((long)dst * AST + rel * 128 + lane * 4) * 2); // byte offset
    asm volatile("global_atomic_pk_add_bf16 %0, %1, %2"
                 :: "v"(off), "v"(d.x), "s"(abase) : "memory");
    asm volatile("global_atomic_pk_add_bf16 %0, %1, %2"
                 :: "v"(off + 4u), "v"(d.y), "s"(abase) : "memory");
    if (lane == 0) atomicAdd(&cnt[(long)dst * NR + rel], 1u);
  }
}

// ---------------- divide agg by max(cnt,1), in place (bf16 dwords) ----------------
__global__ void k_normalize(unsigned short* __restrict__ A, const unsigned* __restrict__ cnt) {
  long tid = (long)blockIdx.x * blockDim.x + threadIdx.x;   // NN*256 dwords
  if (tid >= (long)NN * 256) return;
  long n = tid >> 8; int rem = (int)(tid & 255);
  int r = rem >> 6, dw = rem & 63;
  unsigned c = cnt[(n << 2) + r];
  float inv = 1.0f / (float)(c > 1u ? c : 1u);
  unsigned* p = (unsigned*)A + n * (AST / 2) + (r << 6) + dw;
  unsigned w = *p;
  unsigned lo = f32_bf16(bf16_f32((unsigned short)(w & 0xFFFFu)) * inv);
  unsigned hi = f32_bf16(bf16_f32((unsigned short)(w >> 16)) * inv);
  *p = lo | (hi << 16);
}

// ---- layer-1 GEMM: [N,640]bf16 x [640,128]; B k-slabs staged via async global->LDS ----
__global__ void __launch_bounds__(256) k_gemm1(const unsigned short* __restrict__ A,
                                               const unsigned short* __restrict__ WcatT,
                                               const float* __restrict__ b1,
                                               unsigned short* __restrict__ h) {
  __shared__ unsigned short sB[2][32 * 128];     // double-buffered 8KB B k-slab: [n(128)][kk(32)]
  int tid  = threadIdx.x;
  int lane = tid & 31;
  int wt   = blockIdx.x * 8 + (tid >> 5);
  bool active = wt < (NN / 16);
  int wtc = active ? wt : (NN / 16 - 1);         // clamp: inactive waves still fill LDS + barrier
  int lm = lane & 15, lh = lane >> 4;

  // each thread copies two 16B chunks per slab: 512 chunks = 128 rows x 4 parts
  int c1 = tid, c2 = tid + 256;
  int r1 = c1 >> 2, p1 = c1 & 3;
  int r2 = c2 >> 2, p2 = c2 & 3;
  unsigned long long g1 = (unsigned long long)(uintptr_t)(WcatT + r1 * AST + p1 * 8);
  unsigned long long g2 = (unsigned long long)(uintptr_t)(WcatT + r2 * AST + p2 * 8);
  unsigned l1 = (unsigned)(uintptr_t)&sB[0][r1 * 32 + p1 * 8];
  unsigned l2 = (unsigned)(uintptr_t)&sB[0][r2 * 32 + p2 * 8];
  const unsigned bufstep = 32u * 128u * 2u;      // 8192 B between buffers

  const unsigned short* pa = A + (long)(wtc * 16 + lm) * AST + lh * 8;
  v8f acc[8] = {};

  // prefetch slab 0 into buffer 0 (ASYNCcnt path)
  asm volatile("global_load_async_to_lds_b128 %0, %1, off" :: "v"(l1), "v"(g1) : "memory");
  asm volatile("global_load_async_to_lds_b128 %0, %1, off" :: "v"(l2), "v"(g2) : "memory");

  for (int ks = 0; ks < 20; ks++) {
    int buf = ks & 1;
    if (ks < 19) {                               // kick slab ks+1 into the other buffer
      unsigned long long go = (unsigned long long)((ks + 1) * 64);   // 32 bf16 = 64 B per k-step
      unsigned lo = (buf ^ 1) ? bufstep : 0u;
      asm volatile("global_load_async_to_lds_b128 %0, %1, off" :: "v"(l1 + lo), "v"(g1 + go) : "memory");
      asm volatile("global_load_async_to_lds_b128 %0, %1, off" :: "v"(l2 + lo), "v"(g2 + go) : "memory");
      asm volatile("s_wait_asynccnt 0x2" ::: "memory");  // my slab-ks loads done; 2 newest in flight
    } else {
      asm volatile("s_wait_asynccnt 0x0" ::: "memory");
    }
    __syncthreads();                             // everyone's slab-ks writes visible

    FragBF a;
    a.q[0] = *(const uint4*)(pa + ks * 32);      // K = kb..kb+7
    a.q[1] = *(const uint4*)(pa + ks * 32 + 16); // K = kb+16..kb+23
#pragma unroll
    for (int ct = 0; ct < 8; ct++) {
      const uint4* pb = (const uint4*)&sB[buf][(ct * 16 + lm) * 32 + lh * 8];
      FragBF b;
      b.q[0] = pb[0];                            // K = 8h..8h+7 (ds_load_b128)
      b.q[1] = pb[2];                            // K = 16+8h..16+8h+7
      acc[ct] = __builtin_amdgcn_wmma_f32_16x16x32_bf16(
          false, a.v, false, b.v, (short)0, acc[ct], false, false);
    }
    __syncthreads();                             // reads done before buffer reuse (2 steps later)
  }

  if (!active) return;
  // epilogue: +bias, relu, deterministic dropout (p=0.3), store bf16
  int rbase = wt * 16 + (lh << 3);
  const float inv_keep = 1.0f / 0.7f;
#pragma unroll
  for (int ct = 0; ct < 8; ct++) {
    int col = ct * 16 + lm;
    float bias = b1[col];
#pragma unroll
    for (int v = 0; v < 8; v++) {
      int row = rbase + v;
      float val = acc[ct][v] + bias;
      val = val > 0.0f ? val : 0.0f;
      unsigned s = (unsigned)(row * 128 + col) * 2654435761u;
      s ^= s >> 16; s *= 0x85EBCA6Bu; s ^= s >> 13; s *= 0xC2B2AE35u; s ^= s >> 16;
      val = (s < 3006477107u) ? val * inv_keep : 0.0f;   // keep-prob 0.7
      h[(long)row * 128 + col] = f32_bf16(val);
    }
  }
}

// ---------------- layer-2 GEMM: h[N,128] x Wc2T[16,128]^T -> z f32 [N,16] ----------------
__global__ void __launch_bounds__(256) k_gemm2(const unsigned short* __restrict__ h,
                                               const unsigned short* __restrict__ Wc2T,
                                               float* __restrict__ z) {
  int lane = threadIdx.x & 31;
  int wt = blockIdx.x * (blockDim.x >> 5) + (threadIdx.x >> 5);
  if (wt >= NN / 16) return;
  int lm = lane & 15, lh = lane >> 4;
  const unsigned short* pa = h + (long)(wt * 16 + lm) * 128 + lh * 8;
  const unsigned short* pb = Wc2T + (long)lm * 128 + lh * 8;
  v8f acc = {};
  for (int k0 = 0; k0 < 128; k0 += 32) {
    FragBF a, b;
    a.q[0] = *(const uint4*)(pa + k0);
    a.q[1] = *(const uint4*)(pa + k0 + 16);
    b.q[0] = *(const uint4*)(pb + k0);
    b.q[1] = *(const uint4*)(pb + k0 + 16);
    acc = __builtin_amdgcn_wmma_f32_16x16x32_bf16(
        false, a.v, false, b.v, (short)0, acc, false, false);
  }
  int rbase = wt * 16 + (lh << 3);
#pragma unroll
  for (int v = 0; v < 8; v++) z[(long)(rbase + v) * 16 + lm] = acc[v];
}

// ---------------- layer-2 edge scatter: 2-float messages, f32 atomics ----------------
__global__ void k_scatter2(const int* __restrict__ ei, const int* __restrict__ et,
                           const float* __restrict__ z, float* __restrict__ agg2) {
  long stride = (long)gridDim.x * blockDim.x;
  unsigned long long abase = (unsigned long long)(uintptr_t)agg2;
  for (long e = (long)blockIdx.x * blockDim.x + threadIdx.x; e < NE; e += stride) {
    __builtin_prefetch(ei + e + stride, 0, 1);
    int src = ei[e];
    int dst = ei[NE + e];
    int rel = et[e];
    float2 m = *(const float2*)(z + (long)src * 16 + rel * 2);
    unsigned off = (unsigned)(((long)dst * 8 + rel * 2) * 4);
    asm volatile("global_atomic_add_f32 %0, %1, %2" :: "v"(off), "v"(m.x), "s"(abase) : "memory");
    asm volatile("global_atomic_add_f32 %0, %1, %2" :: "v"(off + 4u), "v"(m.y), "s"(abase) : "memory");
  }
}

// ---------------- final: mean per relation + root part + bias ----------------
__global__ void k_final(const float* __restrict__ agg2, const unsigned* __restrict__ cnt,
                        const float* __restrict__ z, const float* __restrict__ b2,
                        float* __restrict__ out) {
  long n = (long)blockIdx.x * blockDim.x + threadIdx.x;
  if (n >= NN) return;
  float o0 = z[n * 16 + 8] + b2[0];
  float o1 = z[n * 16 + 9] + b2[1];
#pragma unroll
  for (int r = 0; r < NR; r++) {
    unsigned c = cnt[n * NR + r];
    float inv = 1.0f / (float)(c > 1u ? c : 1u);
    o0 += agg2[n * 8 + 2 * r]     * inv;
    o1 += agg2[n * 8 + 2 * r + 1] * inv;
  }
  out[n * 2]     = o0;
  out[n * 2 + 1] = o1;
}

extern "C" void kernel_launch(void* const* d_in, const int* in_sizes, int n_in,
                              void* d_out, int out_size, void* d_ws, size_t ws_size,
                              hipStream_t stream) {
  (void)in_sizes; (void)n_in; (void)out_size; (void)ws_size;
  const float* x      = (const float*)d_in[0];
  const int*   ei     = (const int*)  d_in[1];
  const int*   et     = (const int*)  d_in[2];
  const float* W1     = (const float*)d_in[3];
  const float* root1  = (const float*)d_in[4];
  const float* b1     = (const float*)d_in[5];
  const float* W2     = (const float*)d_in[6];
  const float* root2  = (const float*)d_in[7];
  const float* b2     = (const float*)d_in[8];
  float* out = (float*)d_out;

  // workspace layout (all offsets 256B aligned)
  char* ws = (char*)d_ws;
  unsigned short* A     = (unsigned short*)(ws);                      // 128,000,000 B
  unsigned short* h     = (unsigned short*)(ws + 128000000);          //  25,600,000 B
  float*          z     = (float*)         (ws + 153600000);          //   6,400,000 B
  float*          agg2  = (float*)         (ws + 160000000);          //   3,200,000 B
  unsigned*       cnt   = (unsigned*)      (ws + 163200000);          //   1,600,000 B
  unsigned short* WcatT = (unsigned short*)(ws + 164800000);          //     163,840 B
  unsigned short* Wc2T  = (unsigned short*)(ws + 164963840);          //       4,096 B

  // 1) zero A (agg region + x region overwritten below), and agg2+cnt (contiguous)
  k_zero_u4<<<2048, 256, 0, stream>>>((uint4*)A, 128000000L / 16);
  k_zero_u4<<<512, 256, 0, stream>>>((uint4*)agg2, (3200000L + 1600000L) / 16);
  // 2) weight prep (bf16, transposed for WMMA B fragments)
  k_prep_w1<<<(640 * 128) / 256, 256, 0, stream>>>(W1, root1, WcatT);
  k_prep_w2<<<(16 * 128) / 256, 256, 0, stream>>>(W2, root2, Wc2T);
  // 3) x -> bf16 into A's x-region
  k_convert_x<<<(NN * 64) / 256, 256, 0, stream>>>(x, A);
  // 4) layer-1 edge scatter (pk bf16 atomics, counts)
  k_scatter1<<<4096, 256, 0, stream>>>(ei, et, A, cnt);
  // 5) per-(node,relation) mean
  k_normalize<<<(NN * 256) / 256, 256, 0, stream>>>(A, cnt);
  // 6) layer-1 WMMA GEMM (async LDS-staged B) + bias + relu + dropout -> h (bf16)
  k_gemm1<<<(NN / 16 + 7) / 8, 256, 0, stream>>>(A, WcatT, b1, h);
  // 7) layer-2 WMMA GEMM -> z [N,16] f32 (cols 0..7: per-relation, 8..9: root)
  k_gemm2<<<(NN / 16 + 7) / 8, 256, 0, stream>>>(h, Wc2T, z);
  // 8) layer-2 edge scatter (2-float messages)
  k_scatter2<<<4096, 256, 0, stream>>>(ei, et, z, agg2);
  // 9) final combine
  k_final<<<(NN + 255) / 256, 256, 0, stream>>>(agg2, cnt, z, b2, out);
}
